// LSTMModel_87058987090080
// MI455X (gfx1250) — matrix operand order repaired and checked by reading, not compiled, not run on hardware
//
#include <hip/hip_runtime.h>

// ---------------------------------------------------------------------------
// Fused LSTM (B=4096, T=512, F=16, H=32) + mean-pool + FC for gfx1250.
//
// One wave (32 lanes) owns 16 batch rows. Recurrence computed transposed:
//   z^T[128 gates x 16 batch] = A_wih * x_ext^T + A_whh * h^T   (f16 WMMA, f32 acc)
// Weights preloaded once in WMMA A-layout; bias folded into the input
// projection via an appended ones-row (K=16) in B and a bias column in A.
// Sigmoid gates (i,f,o) have the 0.5 pre-scale of sigmoid(x)=0.5*tanh(0.5x)+0.5
// folded directly into their weight rows, so each gate costs v_tanh + v_fma.
// D-layout of z^T aligns lane/slot-wise with the c/h state tiles, so gate
// nonlinearities and the state update are pure per-lane VALU. Next-step h^T
// B-operand needs only v_cvt_pkrtz + a half-wave __shfl_xor(16).
// ---------------------------------------------------------------------------

typedef __attribute__((ext_vector_type(16))) _Float16 v16h;
typedef __attribute__((ext_vector_type(8)))  float    v8f;
typedef __attribute__((ext_vector_type(2)))  __fp16   pk2;   // matches cvt_pkrtz return type

union U16 {
  v16h v;
  pk2  p[8];
};

// ---- fast activations: hardware TRANS ops, no divide expansion ------------
#if __has_builtin(__builtin_amdgcn_tanhf)
__device__ __forceinline__ float tanh_hw(float x) {
  return __builtin_amdgcn_tanhf(x);
}
// z already carries the 0.5 factor (folded into weights): sigmoid = 0.5*tanh(z)+0.5
__device__ __forceinline__ float sigmoid_prescaled(float zhalf) {
  return __builtin_fmaf(0.5f, __builtin_amdgcn_tanhf(zhalf), 0.5f);
}
#else
__device__ __forceinline__ float tanh_hw(float x) {
  float e = __builtin_amdgcn_exp2f(x * -2.88539008177792681f);
  return __builtin_fmaf(2.0f, __builtin_amdgcn_rcpf(1.0f + e), -1.0f);
}
__device__ __forceinline__ float sigmoid_prescaled(float zhalf) {
  // sigmoid(2*zhalf) = 1/(1 + 2^(-2*zhalf*log2 e))
  float e = __builtin_amdgcn_exp2f(zhalf * -2.88539008177792681f);
  return __builtin_amdgcn_rcpf(1.0f + e);
}
#endif

#define WMMA_F16(A, B, C) \
  __builtin_amdgcn_wmma_f32_16x16x32_f16(false, (A), false, (B), (short)0, (C), false, false)

__global__ __launch_bounds__(32, 1)
void lstm_fused_kernel(const float* __restrict__ x,
                       const float* __restrict__ Wih,   // [128,16]
                       const float* __restrict__ Whh,   // [128,32]
                       const float* __restrict__ bih,   // [128]
                       const float* __restrict__ bhh,   // [128]
                       const float* __restrict__ fcw,   // [32]
                       const float* __restrict__ fcb,   // [1]
                       float* __restrict__ out,         // [B]
                       int T)
{
  constexpr int F_ = 16;
  constexpr int H_ = 32;

  const int  lane  = (int)(threadIdx.x & 31u);
  const int  half  = lane >> 4;          // 0: lanes 0-15, 1: lanes 16-31
  const int  l16   = lane & 15;
  const long bbase = (long)blockIdx.x * 16;

  const pk2 hz = {};

  // ------------------- preload weights in WMMA A-layout -------------------
  // A 16x32 f16 layout: lane m holds row M=m%16;
  //   lanes 0-15 : V0-3 -> K=0..7,  V4-7 -> K=16..23
  //   lanes16-31 : V0-3 -> K=8..15, V4-7 -> K=24..31
  // Gate tiles: 0,1 = i ; 2,3 = f ; 4,5 = g ; 6,7 = o.
  // Sigmoid gates (i,f,o) are pre-scaled by 0.5 (tanh-form sigmoid).
  U16 a_ih[8], a_hh[8];
  {
    const int kbA = half ? 8 : 0;
    #pragma unroll
    for (int gt = 0; gt < 8; ++gt) {
      const float s = (gt == 4 || gt == 5) ? 1.0f : 0.5f;
      const int g = gt * 16 + l16;
      const float* wr = Whh + g * H_;
      #pragma unroll
      for (int j = 0; j < 4; ++j) {
        a_hh[gt].p[j]     = __builtin_amdgcn_cvt_pkrtz(s * wr[kbA + 2*j],
                                                       s * wr[kbA + 2*j + 1]);
        a_hh[gt].p[4 + j] = __builtin_amdgcn_cvt_pkrtz(s * wr[kbA + 16 + 2*j],
                                                       s * wr[kbA + 16 + 2*j + 1]);
      }
      const float* wi = Wih + g * F_;
      #pragma unroll
      for (int j = 0; j < 4; ++j)
        a_ih[gt].p[j] = __builtin_amdgcn_cvt_pkrtz(s * wi[kbA + 2*j],
                                                   s * wi[kbA + 2*j + 1]);
      // K=16 column carries the (scaled) bias, multiplied by ones-row in B.
      const float bias = s * (bih[g] + bhh[g]);
      a_ih[gt].p[4] = half ? hz : __builtin_amdgcn_cvt_pkrtz(bias, 0.0f);
      a_ih[gt].p[5] = hz;
      a_ih[gt].p[6] = hz;
      a_ih[gt].p[7] = hz;
    }
  }

  // ------------------- B operand for x (static part) -----------------------
  // B 32x16 f16 layout (matches A): lanes 0-15 hold col N=n, V0-3 -> K=0..7, V4-7 -> K=16..23;
  //                                lanes16-31 hold col N=n-16, V0-3 -> K=8..15, V4-7 -> K=24..31.
  // Lanes 0-15 carry the constant ones-row at K=16, zeros elsewhere.
  U16 bx;
  #pragma unroll
  for (int j = 0; j < 8; ++j) bx.p[j] = hz;
  if (!half) bx.p[4] = __builtin_amdgcn_cvt_pkrtz(1.0f, 0.0f);

  // ------------------- state ----------------------------------------------
  U16 bh;                                    // h^T in B-layout (f16), starts 0
  #pragma unroll
  for (int j = 0; j < 8; ++j) bh.p[j] = hz;

  float cs0[8], cs1[8], hs0[8], hs1[8];
  #pragma unroll
  for (int r = 0; r < 8; ++r) { cs0[r] = 0.f; cs1[r] = 0.f; hs0[r] = 0.f; hs1[r] = 0.f; }

  const v8f zacc = {};

  // x pointer for this lane's batch row (lanes n and n+16 read the same row;
  // lanes 0-15 use features 0-7, lanes 16-31 use 8-15, keeps EXEC handling trivial and addresses valid).
  const float* xrow = x + ((size_t)(bbase + l16) * (size_t)T) * F_;

  // software pipeline: preload t=0
  float4 q0, q1, q2, q3;
  {
    const float4* xp = (const float4*)xrow;
    q0 = xp[0]; q1 = xp[1]; q2 = xp[2]; q3 = xp[3];
  }

  for (int t = 0; t < T; ++t) {
    // ---- prefetch next timestep's x (issued before WMMA consumption) ----
    const int tn = (t + 1 < T) ? (t + 1) : t;
    const float4* xpn = (const float4*)(xrow + (size_t)tn * F_);
    float4 n0 = xpn[0], n1 = xpn[1], n2 = xpn[2], n3 = xpn[3];

    // ---- build x^T B-operand (lanes 0-15: K=0..7, lanes 16-31: K=8..15; V4-7 keep ones/zeros) --
    if (half == 0) {
      bx.p[0] = __builtin_amdgcn_cvt_pkrtz(q0.x, q0.y);
      bx.p[1] = __builtin_amdgcn_cvt_pkrtz(q0.z, q0.w);
      bx.p[2] = __builtin_amdgcn_cvt_pkrtz(q1.x, q1.y);
      bx.p[3] = __builtin_amdgcn_cvt_pkrtz(q1.z, q1.w);
    } else {
      bx.p[0] = __builtin_amdgcn_cvt_pkrtz(q2.x, q2.y);
      bx.p[1] = __builtin_amdgcn_cvt_pkrtz(q2.z, q2.w);
      bx.p[2] = __builtin_amdgcn_cvt_pkrtz(q3.x, q3.y);
      bx.p[3] = __builtin_amdgcn_cvt_pkrtz(q3.z, q3.w);
    }

    // ---- z^T = Wih_ext * x_ext^T + Whh * h^T  (8 gate tiles of 16x16) ----
    v8f z[8];
    #pragma unroll
    for (int gt = 0; gt < 8; ++gt)
      z[gt] = WMMA_F16(a_ih[gt].v, bx.v, zacc);
    #pragma unroll
    for (int gt = 0; gt < 8; ++gt)
      z[gt] = WMMA_F16(a_hh[gt].v, bh.v, z[gt]);

    // ---- gates + state update (lane-local; tiles 0/1=i, 2/3=f, 4/5=g, 6/7=o)
    float hA[8], hB[8];
    #pragma unroll
    for (int r = 0; r < 8; ++r) {
      {
        const float iv = sigmoid_prescaled(z[0][r]);
        const float fv = sigmoid_prescaled(z[2][r]);
        const float gv = tanh_hw(z[4][r]);
        const float ov = sigmoid_prescaled(z[6][r]);
        const float cn = __builtin_fmaf(fv, cs0[r], iv * gv);
        cs0[r] = cn;
        const float hn = ov * tanh_hw(cn);
        hs0[r] += hn;
        hA[r] = hn;
      }
      {
        const float iv = sigmoid_prescaled(z[1][r]);
        const float fv = sigmoid_prescaled(z[3][r]);
        const float gv = tanh_hw(z[5][r]);
        const float ov = sigmoid_prescaled(z[7][r]);
        const float cn = __builtin_fmaf(fv, cs1[r], iv * gv);
        cs1[r] = cn;
        const float hn = ov * tanh_hw(cn);
        hs1[r] += hn;
        hB[r] = hn;
      }
    }

    // ---- repack h^T into B-layout for next step ----
    // lane n<16 needs H=0..7,16..23 of batch n  : both local (hA, hB)
    // lane n>=16 needs H=8..15,24..31 of batch n: H=0..7 from lane n-16 (sA), H=24..31 local (hB)
    float sA[8], sB[8];
    #pragma unroll
    for (int r = 0; r < 8; ++r) {
      sA[r] = __shfl_xor(hA[r], 16, 32);
      sB[r] = __shfl_xor(hB[r], 16, 32);
    }
    #pragma unroll
    for (int j = 0; j < 4; ++j) {
      const float l0 = half ? sA[2*j]     : hA[2*j];
      const float l1 = half ? sA[2*j + 1] : hA[2*j + 1];
      const float u0 = hB[2*j];
      const float u1 = hB[2*j + 1];
      bh.p[j]     = __builtin_amdgcn_cvt_pkrtz(l0, l1);
      bh.p[4 + j] = __builtin_amdgcn_cvt_pkrtz(u0, u1);
    }

    q0 = n0; q1 = n1; q2 = n2; q3 = n3;
  }

  // ------------------- mean pool + FC(32 -> 1) -----------------------------
  // hs tile jt, slot r at this lane corresponds to H = 16*jt + r + 8*half.
  const int ho = half ? 8 : 0;
  float part = 0.0f;
  #pragma unroll
  for (int r = 0; r < 8; ++r) {
    part += hs0[r] * fcw[r + ho];
    part += hs1[r] * fcw[16 + r + ho];
  }
  part += __shfl_xor(part, 16, 32);
  if (half == 0)
    out[bbase + l16] = part * (1.0f / (float)T) + fcb[0];
}

extern "C" void kernel_launch(void* const* d_in, const int* in_sizes, int n_in,
                              void* d_out, int out_size, void* d_ws, size_t ws_size,
                              hipStream_t stream) {
  const float* x   = (const float*)d_in[0];
  const float* Wih = (const float*)d_in[1];
  const float* Whh = (const float*)d_in[2];
  const float* bih = (const float*)d_in[3];
  const float* bhh = (const float*)d_in[4];
  const float* fcw = (const float*)d_in[5];
  const float* fcb = (const float*)d_in[6];
  float* out = (float*)d_out;

  const int T = 512;
  const int F = 16;
  const int B = in_sizes[0] / (T * F);   // 4096

  dim3 grid(B / 16);
  dim3 block(32);
  hipLaunchKernelGGL(lstm_fused_kernel, grid, block, 0, stream,
                     x, Wih, Whh, bih, bhh, fcw, fcb, out, T);
}
